// BasicNCAModel_21638045237333
// MI455X (gfx1250) — matrix operand order repaired and checked by reading, not compiled, not run on hardware
//
#include <hip/hip_runtime.h>
#include <stdint.h>

typedef _Float16 v16h __attribute__((ext_vector_type(16)));
typedef _Float16 v8h  __attribute__((ext_vector_type(8)));
typedef float    v8f  __attribute__((ext_vector_type(8)));
typedef unsigned int uint32x4 __attribute__((ext_vector_type(4)));
typedef int          int32x4  __attribute__((ext_vector_type(4)));
typedef int          int32x8  __attribute__((ext_vector_type(8)));

#define Bv   16
#define Hv   128
#define Wv   128
#define Cv   64
#define FAN  192
#define HIDv 256

#define WHB_HALVES (6 * 16 * 32 * 16)     // 49152 f16 = 96 KB
#define WFB_HALVES (8 * 4 * 32 * 16)      // 16384 f16 = 32 KB

__device__ __forceinline__ int refl(int i) { return i < 0 ? -i : (i > 127 ? 254 - i : i); }

// Low 32 bits of a flat pointer into LDS == workgroup-relative LDS byte address.
__device__ __forceinline__ unsigned lds_addr32(const void* p) {
    return (unsigned)(uintptr_t)p;
}

__device__ __forceinline__ void async_copy_b128(unsigned lds_dst, const void* gsrc) {
    asm volatile("global_load_async_to_lds_b128 %0, %1, off"
                 :: "v"(lds_dst), "v"((unsigned long long)(uintptr_t)gsrc) : "memory");
}

// ---- one-time: pre-swizzle f32 weights into WMMA-B-layout f16 blobs in d_ws ----
__global__ void prep_weights_kernel(const float* __restrict__ whid,
                                    const float* __restrict__ wfin,
                                    _Float16* __restrict__ wsWhB,
                                    _Float16* __restrict__ wsWfB)
{
    const int tid = blockIdx.x * blockDim.x + threadIdx.x;
    const int nth = gridDim.x * blockDim.x;
    for (int idx = tid; idx < HIDv * FAN; idx += nth) {
        int o = idx / FAN, f = idx - o * FAN;        // whid is [o][f] row-major
        int kc = f >> 5, r = f & 31;
        int lh = (r >> 3) & 1, v = ((r >> 4) * 8 + (r & 7)) >> 1, hl = r & 1;
        int j = o >> 4, nc = o & 15;
        wsWhB[((kc * 16 + j) * 32 + lh * 16 + nc) * 16 + v * 2 + hl] = (_Float16)whid[idx];
    }
    for (int idx = tid; idx < Cv * HIDv; idx += nth) {
        int c = idx >> 8, o = idx & 255;             // wfin is [c][o] row-major
        int kc2 = o >> 5, r = o & 31;
        int lh = (r >> 3) & 1, v = ((r >> 4) * 8 + (r & 7)) >> 1, hl = r & 1;
        int n2 = c >> 4, nc = c & 15;
        wsWfB[(((kc2 * 4 + n2) * 32) + lh * 16 + nc) * 16 + v * 2 + hl] = (_Float16)wfin[idx];
    }
}

__launch_bounds__(128, 1)
__global__ void nca_step_kernel(const float* __restrict__ xin,
                                const float* __restrict__ wc1,
                                const float* __restrict__ wc2,
                                const _Float16* __restrict__ wsWhB,
                                const float* __restrict__ bhid,
                                const _Float16* __restrict__ wsWfB,
                                const float* __restrict__ rnd,
                                float* __restrict__ xout,
                                int step)
{
    // Union region (50688 B): first the f32 x tile [3][66][64] (async-DMA'd),
    // later overwritten (TDM) with the w_final WMMA-B blob [8][4][32][16] f16.
    __shared__ __align__(64) unsigned char sUni[3 * 66 * 64 * 4];
    __shared__ __align__(64) _Float16 sWhB[WHB_HALVES];
    __shared__ float sBias[HIDv];
    __shared__ float sWc[2 * 64 * 9];
    // per-wave h staging in WMMA-A layout: [wave][pixel-row(16)][k(32)]
    __shared__ __align__(64) _Float16 sHa[4][16][32];

    float    (*sXf)[66][64] = (float (*)[66][64])sUni;
    _Float16* sWfB          = (_Float16*)sUni;

    const int tid   = threadIdx.x;
    const int lane  = tid & 31;
    const int wv    = tid >> 5;
    const int mrow  = lane & 15;   // WMMA row / N-col within tile
    const int khalf = lane >> 4;   // lane-half for A/B K offsets

    const int seg  = blockIdx.x;
    const int wseg = seg & 1;
    const int h    = (seg >> 1) & (Hv - 1);
    const int b    = seg >> 8;
    const int col0 = wseg * 64;

    // ---- async DMA: pre-swizzled w_hidden blob (96 KB, linear copy) ----
    {
        const unsigned ldsbase = lds_addr32(sWhB);
        for (int idx = tid; idx < (WHB_HALVES * 2) / 16; idx += 128)    // 6144 x b128
            async_copy_b128(ldsbase + idx * 16, (const char*)wsWhB + idx * 16);
    }
    // ---- async DMA: x tile (3 rows x 66 halo cols x 64 ch, f32) ----
    for (int dy = 0; dy < 3; ++dy) {
        const int gh = refl(h - 1 + dy);
        const float* rowp = xin + (size_t)(b * Hv + gh) * Wv * Cv;
        for (int t = tid; t < 66 * 16; t += 128) {
            int col = t >> 4, ci = t & 15;
            int gw = refl(col0 - 1 + col);
            async_copy_b128(lds_addr32(&sXf[dy][col][ci * 4]), rowp + gw * Cv + ci * 4);
        }
    }
    // ---- small fills (regular loads, overlap with async DMA) ----
    for (int idx = tid; idx < HIDv; idx += 128) sBias[idx] = bhid[idx];
    for (int idx = tid; idx < 576;  idx += 128) { sWc[idx] = wc1[idx]; sWc[576 + idx] = wc2[idx]; }

#if __has_builtin(__builtin_amdgcn_s_wait_asynccnt)
    __builtin_amdgcn_s_wait_asynccnt(0);
#else
    asm volatile("s_wait_asynccnt 0" ::: "memory");
#endif
    __syncthreads();

    // ---- build A operand in registers: 16 pixels x 192 features (f16) ----
    // feature f: [0,64)=x, [64,128)=dwconv1, [128,192)=dwconv2
    const int lx = wv * 16 + mrow + 1;   // col in sXf (halo offset)
    v16h areg[6];
#pragma unroll
    for (int kc = 0; kc < 6; ++kc) {
        const int g = kc >> 1;           // feature group is uniform per kc
#pragma unroll
        for (int s = 0; s < 16; ++s) {
            const int kmap = (s < 8) ? (khalf * 8 + s) : (16 + khalf * 8 + (s - 8));
            const int c = (kc & 1) * 32 + kmap;
            float val;
            if (g == 0) {
                val = sXf[1][lx][c];
            } else {
                const float* wk = &sWc[(g - 1) * 576 + c * 9];
                float acc = 0.f;
#pragma unroll
                for (int kx = 0; kx < 3; ++kx)
#pragma unroll
                    for (int ky = 0; ky < 3; ++ky)
                        acc += wk[kx * 3 + ky] * sXf[ky][lx - 1 + kx][c];
                val = acc;
            }
            areg[kc][s] = (_Float16)val;
        }
    }

    // ---- x tile dead: TDM the w_final blob (32 KB) into the union region ----
    __syncthreads();
#if __has_builtin(__builtin_amdgcn_tensor_load_to_lds)
    if (wv == 0) {
        const unsigned long long ga = (unsigned long long)(uintptr_t)wsWfB;
        uint32x4 g0;
        g0[0] = 1u;                                              // count=1 valid descriptor
        g0[1] = lds_addr32(sWfB);                                // lds_addr
        g0[2] = (unsigned)(ga & 0xFFFFFFFFu);                    // global_addr[31:0]
        g0[3] = (unsigned)((ga >> 32) & 0x1FFFFFFu) | 0x80000000u; // ga[56:32] | type=2
        int32x8 g1;
        g1[0] = 0x00020000;                 // data_size=2 -> 4-byte elements
        g1[1] = (int)(8192u << 16);         // tensor_dim0 = 8192 dwords (lo16 in [31:16])
        g1[2] = (int)(1u << 16);            // tensor_dim0 hi=0, tensor_dim1 = 1
        g1[3] = (int)(8192u << 16);         // tensor_dim1 hi=0, tile_dim0 = 8192
        g1[4] = 1;                          // tile_dim1 = 1, tile_dim2 = 0
        g1[5] = 8192;                       // tensor_dim0_stride lo32
        g1[6] = 0;                          // stride hi, tensor_dim1_stride lo16
        g1[7] = 0;
        int32x4 z4 = {0, 0, 0, 0};
#if defined(__clang_major__) && (__clang_major__ >= 23)
        int32x8 z8 = {0, 0, 0, 0, 0, 0, 0, 0};
        __builtin_amdgcn_tensor_load_to_lds(g0, g1, z4, z4, z8, 0);
#else
        __builtin_amdgcn_tensor_load_to_lds(g0, g1, z4, z4, 0);
#endif
        __builtin_amdgcn_s_wait_tensorcnt((short)0);
    }
#else
    {   // fallback: async b128 linear copy
        const unsigned ldsbase = lds_addr32(sWfB);
        for (int idx = tid; idx < (WFB_HALVES * 2) / 16; idx += 128)
            async_copy_b128(ldsbase + idx * 16, (const char*)wsWfB + idx * 16);
#if __has_builtin(__builtin_amdgcn_s_wait_asynccnt)
        __builtin_amdgcn_s_wait_asynccnt(0);
#else
        asm volatile("s_wait_asynccnt 0" ::: "memory");
#endif
    }
#endif
    __syncthreads();

    // ---- fused GEMM1 (192->256, ReLU) and GEMM2 (256->64) ----
    v8f dxacc[4] = {v8f{}, v8f{}, v8f{}, v8f{}};

    for (int j = 0; j < 16; ++j) {
        v8f acc = {};
#pragma unroll
        for (int kc = 0; kc < 6; ++kc) {
            const v16h bt = *(const v16h*)&sWhB[(((kc * 16 + j) * 32) + lane) * 16];
            acc = __builtin_amdgcn_wmma_f32_16x16x32_f16(
                false, areg[kc], false, bt, (short)0, acc, false, false);
        }
        const float bo = sBias[j * 16 + mrow];
#pragma unroll
        for (int i = 0; i < 8; ++i) {
            float hv = acc[i] + bo;
            hv = hv > 0.f ? hv : 0.f;
            sHa[wv][khalf * 8 + i][(j & 1) * 16 + mrow] = (_Float16)hv;
        }
        if (j & 1) {
            asm volatile("s_wait_dscnt 0" ::: "memory");
            const int jp = j >> 1;                 // K-chunk of GEMM2 (o in [jp*32, jp*32+32))
            v16h a2;
            {
                const _Float16* hp = &sHa[wv][mrow][0];
                const v8h lo = *(const v8h*)(hp + khalf * 8);
                const v8h hi = *(const v8h*)(hp + 16 + khalf * 8);
#pragma unroll
                for (int s = 0; s < 8; ++s) { a2[s] = lo[s]; a2[8 + s] = hi[s]; }
            }
#pragma unroll
            for (int n2 = 0; n2 < 4; ++n2) {
                const v16h bt2 = *(const v16h*)&sWfB[(((jp * 4 + n2) * 32) + lane) * 16];
                dxacc[n2] = __builtin_amdgcn_wmma_f32_16x16x32_f16(
                    false, a2, false, bt2, (short)0, dxacc[n2], false, false);
            }
        }
    }

    // ---- epilogue: gate, channel mask, life mask, write NHWC ----
    const int pixg = ((b * Hv + h) * Wv) + col0 + wv * 16;
    float gate[8];
    int prow[8];
#pragma unroll
    for (int i = 0; i < 8; ++i) {
        const int M = khalf * 8 + i;               // pixel row in tile
        prow[i] = pixg + M;
        const float rv = rnd[((step * Bv + b) * Hv + h) * Wv + col0 + wv * 16 + M];
        gate[i] = rv > 0.5f ? 1.f : 0.f;
    }
    float life[8];
    {   // n2 = 0 tile first: contains channel 0 (life source)
        const int c = mrow;
        const float chm = (c >= 4) ? 1.f : 0.f;    // image channels immutable
#pragma unroll
        for (int i = 0; i < 8; ++i) {
            const float xo = xin[prow[i] * Cv + c];
            const float xn = xo + dxacc[0][i] * gate[i] * chm;
            const int srcl = khalf << 4;           // lane holding channel 0 of this row
            const float c0 = __int_as_float(
                __builtin_amdgcn_ds_bpermute(srcl << 2, __float_as_int(xn)));
            life[i] = c0 > 0.f ? 1.f : 0.f;
            xout[prow[i] * Cv + c] = xn * life[i];
        }
    }
#pragma unroll
    for (int n2 = 1; n2 < 4; ++n2) {
        const int c = n2 * 16 + mrow;              // c >= 16, mask always 1
#pragma unroll
        for (int i = 0; i < 8; ++i) {
            const float xo = xin[prow[i] * Cv + c];
            const float xn = xo + dxacc[n2][i] * gate[i];
            xout[prow[i] * Cv + c] = xn * life[i];
        }
    }
}

extern "C" void kernel_launch(void* const* d_in, const int* in_sizes, int n_in,
                              void* d_out, int out_size, void* d_ws, size_t ws_size,
                              hipStream_t stream) {
    (void)in_sizes; (void)n_in; (void)out_size; (void)ws_size;
    const float* x    = (const float*)d_in[0];
    const float* wc1  = (const float*)d_in[1];
    const float* wc2  = (const float*)d_in[2];
    const float* whid = (const float*)d_in[3];
    const float* bhid = (const float*)d_in[4];
    const float* wfin = (const float*)d_in[5];
    const float* rnd  = (const float*)d_in[6];

    float* bufA = (float*)d_ws;                           // 64 MiB state scratch
    char*  wsb  = (char*)d_ws + (size_t)Bv * Hv * Wv * Cv * 4;
    _Float16* wsWhB = (_Float16*)wsb;                     // 96 KB swizzled w_hidden
    _Float16* wsWfB = (_Float16*)(wsb + WHB_HALVES * 2);  // 32 KB swizzled w_final
    float* bufB = (float*)d_out;

    prep_weights_kernel<<<dim3(64), dim3(256), 0, stream>>>(whid, wfin, wsWhB, wsWfB);

    dim3 grid(Bv * Hv * 2), block(128);
    for (int s = 0; s < 8; ++s) {
        const float* src = (s == 0) ? x : ((s & 1) ? bufA : bufB);
        float* dst = (s & 1) ? bufB : bufA;               // s=7 writes bufB == d_out
        nca_step_kernel<<<grid, block, 0, stream>>>(src, wc1, wc2, wsWhB, bhid,
                                                    wsWfB, rnd, dst, s);
    }
}
